// Struct2SeqLO_10204842295365
// MI455X (gfx1250) — compile-verified, matrix operand order checked by reading, not run on hardware
//
#include <hip/hip_runtime.h>
#include <hip/hip_bf16.h>
#include <math.h>

// ---------------- model dims ----------------
#define BB   8
#define NN   512
#define KNB  30
#define HID  128
#define RN   (BB*NN)        // 4096 node rows
#define RE   (BB*NN*KNB)    // 122880 edge rows

typedef _Float16 half_t;
typedef __attribute__((ext_vector_type(16))) _Float16 v16h;
typedef __attribute__((ext_vector_type(8)))  _Float16 v8h;
typedef __attribute__((ext_vector_type(8)))  float    v8f;

// ---------------- small device helpers ----------------
__device__ __forceinline__ float sgnf(float x){ return (x>0.f)?1.f:((x<0.f)?-1.f:0.f); }
__device__ __forceinline__ float wred_sum(float v){
  #pragma unroll
  for (int o=16;o;o>>=1) v += __shfl_xor(v,o,32);
  return v;
}
__device__ __forceinline__ float wred_max(float v){
  #pragma unroll
  for (int o=16;o;o>>=1) v = fmaxf(v,__shfl_xor(v,o,32));
  return v;
}
__device__ __forceinline__ void cross3(const float* a, const float* b, float* c){
  c[0]=a[1]*b[2]-a[2]*b[1];
  c[1]=a[2]*b[0]-a[0]*b[2];
  c[2]=a[0]*b[1]-a[1]*b[0];
}
__device__ __forceinline__ void norm3(float* v){
  float n = sqrtf(v[0]*v[0]+v[1]*v[1]+v[2]*v[2]);
  n = fmaxf(n, 1e-12f);
  v[0]/=n; v[1]/=n; v[2]/=n;
}
__device__ __forceinline__ void diffnorm(const float* b, const float* a, float* u){
  u[0]=b[0]-a[0]; u[1]=b[1]-a[1]; u[2]=b[2]-a[2];
  norm3(u);
}
__device__ __forceinline__ void loadXb(const float* X, int b, int i, float* p){
  size_t base = (((size_t)b*NN + (size_t)(i/3))*4 + (size_t)(i%3))*3;
  p[0]=X[base]; p[1]=X[base+1]; p[2]=X[base+2];
}
__device__ __forceinline__ void loadXca(const float* X, int n_, int b, float* p){
  (void)0;
}
__device__ __forceinline__ void loadXcaBN(const float* X, int b, int n, float* p){
  size_t base = (((size_t)b*NN + (size_t)n)*4 + 1)*3;
  p[0]=X[base]; p[1]=X[base+1]; p[2]=X[base+2];
}

// ---------------- distances + top-K neighbors ----------------
__global__ __launch_bounds__(32) void dist_topk_kernel(const float* __restrict__ X,
                                                       float* __restrict__ Dnb,
                                                       int* __restrict__ Eidx){
  __shared__ float Drow[NN];
  const int bi = blockIdx.x;
  const int b  = bi / NN, i = bi % NN;
  const int lane = threadIdx.x;
  float xi[3]; loadXcaBN(X, b, i, xi);
  for (int j = lane; j < NN; j += 32){
    float xj[3]; loadXcaBN(X, b, j, xj);
    float dx = xi[0]-xj[0], dy = xi[1]-xj[1], dz = xi[2]-xj[2];
    Drow[j] = sqrtf(dx*dx+dy*dy+dz*dz + 1e-6f);
  }
  __syncthreads();
  for (int s = 0; s < KNB; ++s){
    float bv = 3.0e38f; int bj = NN;
    for (int j = lane; j < NN; j += 32){
      float v = Drow[j];
      if (v < bv){ bv = v; bj = j; }
    }
    #pragma unroll
    for (int o=16;o;o>>=1){
      float ov = __shfl_xor(bv,o,32);
      int   oj = __shfl_xor(bj,o,32);
      if (ov < bv || (ov == bv && oj < bj)){ bv = ov; bj = oj; }
    }
    if (lane == 0){
      Dnb[(size_t)bi*KNB + s]  = bv;
      Eidx[(size_t)bi*KNB + s] = bj;
      Drow[bj] = 3.0e38f;
    }
    __syncthreads();
  }
}

// ---------------- dihedral node features (B,N,6) ----------------
__global__ void dihedrals_kernel(const float* __restrict__ X, float* __restrict__ V6){
  int t = blockIdx.x*blockDim.x + threadIdx.x;
  if (t >= RN) return;
  int b = t / NN, n = t % NN;
  #pragma unroll
  for (int a = 0; a < 3; ++a){
    int p = 3*n + a;
    float Dv = 0.f;
    if (p >= 1 && p <= 3*NN-3){
      int tp = p - 1;
      float p0[3],p1[3],p2[3],p3[3];
      loadXb(X,b,tp,p0); loadXb(X,b,tp+1,p1); loadXb(X,b,tp+2,p2); loadXb(X,b,tp+3,p3);
      float u2[3],u1[3],u0[3];
      diffnorm(p1,p0,u2); diffnorm(p2,p1,u1); diffnorm(p3,p2,u0);
      float n2[3],n1[3];
      cross3(u2,u1,n2); norm3(n2);
      cross3(u1,u0,n1); norm3(n1);
      float cosD = n2[0]*n1[0]+n2[1]*n1[1]+n2[2]*n1[2];
      cosD = fminf(fmaxf(cosD, -1.f+1e-7f), 1.f-1e-7f);
      float sg = sgnf(u2[0]*n1[0]+u2[1]*n1[1]+u2[2]*n1[2]);
      Dv = sg * acosf(cosD);
    }
    V6[(size_t)t*6 + a]     = cosf(Dv);
    V6[(size_t)t*6 + 3 + a] = sinf(Dv);
  }
}

// ---------------- local frames O (B,N,9) ----------------
__global__ void orient_kernel(const float* __restrict__ X, float* __restrict__ O9){
  int t = blockIdx.x*blockDim.x + threadIdx.x;
  if (t >= RN) return;
  int b = t / NN, n = t % NN;
  float* o = O9 + (size_t)t*9;
  if (n >= 1 && n <= NN-3){
    int r = n - 1;
    float c0[3],c1[3],c2[3];
    loadXcaBN(X,b,r,c0); loadXcaBN(X,b,r+1,c1); loadXcaBN(X,b,r+2,c2);
    float u2[3],u1[3];
    diffnorm(c1,c0,u2); diffnorm(c2,c1,u1);
    float o1[3] = {u2[0]-u1[0], u2[1]-u1[1], u2[2]-u1[2]};
    norm3(o1);
    float cr[3]; cross3(o1,u1,cr);
    o[0]=o1[0]; o[1]=o1[1]; o[2]=o1[2];
    o[3]=cr[0]; o[4]=cr[1]; o[5]=cr[2];
    o[6]=u1[0]; o[7]=u1[1]; o[8]=u1[2];
  } else {
    #pragma unroll
    for (int i=0;i<9;i++) o[i]=0.f;
  }
}

// ---------------- node embed: V6 @ Wn + bn -> LN (B,N,128) ----------------
__global__ __launch_bounds__(32) void node_embed_kernel(const float* __restrict__ V6,
    const float* __restrict__ Wn, const float* __restrict__ bn_,
    const float* __restrict__ gn, const float* __restrict__ betan,
    float* __restrict__ Vn){
  int t = blockIdx.x, lane = threadIdx.x;
  float f[6];
  #pragma unroll
  for (int c=0;c<6;c++) f[c] = V6[(size_t)t*6 + c];
  float v[4];
  #pragma unroll
  for (int i=0;i<4;i++){
    int j = lane + 32*i;
    float acc = bn_[j];
    #pragma unroll
    for (int c=0;c<6;c++) acc += f[c]*Wn[c*128 + j];
    v[i]=acc;
  }
  float mu = wred_sum(v[0]+v[1]+v[2]+v[3]) * (1.f/128.f);
  float s2 = 0.f;
  #pragma unroll
  for (int i=0;i<4;i++){ float d=v[i]-mu; s2 += d*d; }
  float inv = rsqrtf(wred_sum(s2)*(1.f/128.f) + 1e-6f);
  #pragma unroll
  for (int i=0;i<4;i++){
    int j = lane + 32*i;
    Vn[(size_t)t*128 + j] = gn[j]*(v[i]-mu)*inv + betan[j];
  }
}

// ---------------- edge features -> @We + LN (B,N,K,128) f32 ----------------
__global__ __launch_bounds__(32) void edge_feat_kernel(const float* __restrict__ X,
    const float* __restrict__ Dnb, const int* __restrict__ Eidx,
    const float* __restrict__ O9,
    const float* __restrict__ We, const float* __restrict__ be,
    const float* __restrict__ ge, const float* __restrict__ betae,
    float* __restrict__ Efeat){
  int e = blockIdx.x, lane = threadIdx.x;
  int n = (e / KNB) % NN;
  int b =  e / (KNB * NN);
  int idx = Eidx[e];
  float Dv = Dnb[e];
  float feat[39];
  float d = (float)(idx - n);
  #pragma unroll
  for (int i=0;i<8;i++){
    float fr = expf(-0.5756462732f * (float)(2*i));  // ln(10000)/16
    float ang = d*fr;
    feat[i]   = cosf(ang);
    feat[8+i] = sinf(ang);
  }
  #pragma unroll
  for (int j=0;j<16;j++){
    float mu = (20.f/15.f)*(float)j;
    float tt = (Dv - mu)*(1.f/1.25f);
    feat[16+j] = expf(-tt*tt);
  }
  float Om[9], On[9];
  #pragma unroll
  for (int i=0;i<9;i++){
    Om[i]=O9[((size_t)b*NN+n)*9+i];
    On[i]=O9[((size_t)b*NN+idx)*9+i];
  }
  float xi[3], xj[3];
  loadXcaBN(X,b,n,xi); loadXcaBN(X,b,idx,xj);
  float dX[3] = {xj[0]-xi[0], xj[1]-xi[1], xj[2]-xi[2]};
  float dU[3];
  #pragma unroll
  for (int i=0;i<3;i++) dU[i] = Om[i*3]*dX[0] + Om[i*3+1]*dX[1] + Om[i*3+2]*dX[2];
  norm3(dU);
  float R[9];   // R[i*3+l] = sum_j Om[j][i] * On[j][l]
  #pragma unroll
  for (int i=0;i<3;i++)
    #pragma unroll
    for (int l=0;l<3;l++)
      R[i*3+l] = Om[0*3+i]*On[0*3+l] + Om[1*3+i]*On[1*3+l] + Om[2*3+i]*On[2*3+l];
  float Rxx=R[0], Ryy=R[4], Rzz=R[8];
  float m0 = 0.5f*sqrtf(fabsf(1.f + Rxx - Ryy - Rzz));
  float m1 = 0.5f*sqrtf(fabsf(1.f - Rxx + Ryy - Rzz));
  float m2 = 0.5f*sqrtf(fabsf(1.f - Rxx - Ryy + Rzz));
  float qx = sgnf(R[2*3+1]-R[1*3+2])*m0;
  float qy = sgnf(R[0*3+2]-R[2*3+0])*m1;
  float qz = sgnf(R[1*3+0]-R[0*3+1])*m2;
  float qw = 0.5f*sqrtf(fmaxf(0.f, 1.f + Rxx + Ryy + Rzz));
  float qn = fmaxf(sqrtf(qx*qx+qy*qy+qz*qz+qw*qw), 1e-12f);
  feat[32]=dU[0]; feat[33]=dU[1]; feat[34]=dU[2];
  feat[35]=qx/qn; feat[36]=qy/qn; feat[37]=qz/qn; feat[38]=qw/qn;
  float v[4];
  #pragma unroll
  for (int i=0;i<4;i++){
    int j = lane + 32*i;
    float acc = be[j];
    for (int f=0; f<39; ++f) acc += feat[f]*We[f*128 + j];
    v[i]=acc;
  }
  float mu = wred_sum(v[0]+v[1]+v[2]+v[3]) * (1.f/128.f);
  float s2 = 0.f;
  #pragma unroll
  for (int i=0;i<4;i++){ float dd=v[i]-mu; s2 += dd*dd; }
  float inv = rsqrtf(wred_sum(s2)*(1.f/128.f) + 1e-6f);
  #pragma unroll
  for (int i=0;i<4;i++){
    int j = lane + 32*i;
    Efeat[(size_t)e*128 + j] = ge[j]*(v[i]-mu)*inv + betae[j];
  }
}

// ---------------- weight pre-swizzle: f32 (Kdim x Nout) -> fragment-ready f16 ----------------
// layout: [n_tile][k_tile][lane(32)][16 halves contiguous]
// lane L<16  -> col n_tile*16+L, k = {0..7, 16..23} of k_tile
// lane L+16  -> col n_tile*16+L, k = {8..15, 24..31} of k_tile
__global__ void swizzle_w_kernel(const float* __restrict__ W, half_t* __restrict__ Wh,
                                 int Kdim, int Nout){
  int t = blockIdx.x*blockDim.x + threadIdx.x;
  int total = Kdim*Nout;
  if (t >= total) return;
  int k = t / Nout, c = t % Nout;
  int tn = c >> 4, lc = c & 15;
  int tk = k >> 5, ko = k & 31;
  int lane = lc + (((ko>>3)&1) << 4);
  int idx  = (ko & 7) + ((ko >> 4) << 3);
  Wh[(((size_t)tn*(Kdim>>5) + tk)*32 + lane)*16 + idx] = (half_t)W[t];
}

// ---------------- WMMA GEMM: 64 rows x 128 cols per block, 4 acc per wave ----------------
// mode 0: A0 rows of length Kdim (a0_half selects f32/f16; relu_in optional)
// mode 1: [h_E.f16(128) | gather(A1.f32)(128)]                       (encoder h_EV)
// mode 2: [h_E.f16(128) | ar?gather(A1):0 | gather(ar?A2:A2b)]       (decoder blend)
__global__ __launch_bounds__(256) void gemm_wmma_kernel(
    const void* __restrict__ A0v, int a0_half,
    const float* __restrict__ A1, const float* __restrict__ A2,
    const float* __restrict__ A2b, const int* __restrict__ Eidx,
    const half_t* __restrict__ Wsw, const float* __restrict__ bias,
    void* __restrict__ outv, int out_half,
    int rows, int Kdim, int Nout, int mode, int relu_in){
  __shared__ alignas(16) half_t lA[64*32];
  __shared__ int sIdx[64];
  __shared__ int sAr[64];
  const int tid  = threadIdx.x;
  const int wave = tid >> 5, lane = tid & 31;
  const int row0 = blockIdx.x * 64;
  const int col0 = blockIdx.y * 128;
  const int hh = lane >> 4, l = lane & 15;
  if (mode != 0 && tid < 64){
    int r = row0 + tid;
    int nn = (r / KNB) % NN;
    int idx = Eidx[r];
    sIdx[tid] = idx;
    sAr[tid]  = (idx < nn) ? 1 : 0;
  }
  __syncthreads();
  v8f acc0 = {}, acc1 = {}, acc2 = {}, acc3 = {};
  const int ktiles = Kdim >> 5;
  const int tn = blockIdx.y*8 + wave;   // global 16-col tile id
  auto loadA = [&](int s)->v16h{
    const half_t* ap = &lA[(s*16 + l)*32 + hh*8];
    v8h lo = *(const v8h*)ap;
    v8h hi = *(const v8h*)(ap + 16);
    v16h a;
    #pragma unroll
    for (int i=0;i<8;i++){ a[i]=lo[i]; a[8+i]=hi[i]; }
    return a;
  };
  for (int kt = 0; kt < ktiles; ++kt){
    const int ks = kt << 5;
    // ---- stage A tile (64 x 32) with fused gather/concat + f32->f16
    for (int e = tid; e < 64*32; e += 256){
      int m = e >> 5, c = (e & 31) + ks;
      int r = row0 + m;
      float v;
      if (mode == 0){
        v = a0_half ? (float)((const half_t*)A0v)[(size_t)r*Kdim + c]
                    : ((const float*)A0v)[(size_t)r*Kdim + c];
      } else {
        int idx = sIdx[m];
        int b = r / (KNB*NN);
        if (c < 128){
          v = (float)((const half_t*)A0v)[(size_t)r*128 + c];
        } else if (c < 256){
          if (mode == 1) v = A1[((size_t)b*NN + idx)*128 + (c-128)];
          else           v = sAr[m] ? A1[((size_t)b*NN + idx)*128 + (c-128)] : 0.f;
        } else {
          const float* s = sAr[m] ? A2 : A2b;
          v = s[((size_t)b*NN + idx)*128 + (c-256)];
        }
      }
      if (relu_in) v = fmaxf(v, 0.f);
      lA[e] = (half_t)v;
    }
    __syncthreads();
    // ---- B fragment direct from swizzled weights (2 contiguous b128 per lane)
    const half_t* bp = Wsw + (((size_t)tn*ktiles + kt)*32 + lane)*16;
    v16h bf;
    {
      v8h blo = *(const v8h*)bp;
      v8h bhi = *(const v8h*)(bp + 8);
      #pragma unroll
      for (int i=0;i<8;i++){ bf[i]=blo[i]; bf[8+i]=bhi[i]; }
    }
    if (kt + 1 < ktiles) __builtin_prefetch(bp + 512, 0, 1);
    // ---- 4 row sub-tiles share the B fragment
    v16h a0f = loadA(0);
    acc0 = __builtin_amdgcn_wmma_f32_16x16x32_f16(false, a0f, false, bf, (short)0, acc0, false, false);
    v16h a1f = loadA(1);
    acc1 = __builtin_amdgcn_wmma_f32_16x16x32_f16(false, a1f, false, bf, (short)0, acc1, false, false);
    v16h a2f = loadA(2);
    acc2 = __builtin_amdgcn_wmma_f32_16x16x32_f16(false, a2f, false, bf, (short)0, acc2, false, false);
    v16h a3f = loadA(3);
    acc3 = __builtin_amdgcn_wmma_f32_16x16x32_f16(false, a3f, false, bf, (short)0, acc3, false, false);
    __syncthreads();
  }
  const int col = col0 + wave*16 + l;
  const float bz = bias ? bias[col] : 0.f;
  v8f accs[4] = {acc0, acc1, acc2, acc3};
  #pragma unroll
  for (int s=0;s<4;s++){
    #pragma unroll
    for (int i=0;i<8;i++){
      int m = s*16 + hh*8 + i;
      float val = accs[s][i] + bz;
      if (out_half) ((half_t*)outv)[(size_t)(row0+m)*Nout + col] = (half_t)val;
      else          ((float*) outv)[(size_t)(row0+m)*Nout + col] = val;
    }
  }
}

// ---------------- neighbor attention (block per node, wave per head) ----------------
__global__ __launch_bounds__(128) void attn_kernel(const float* __restrict__ Q,
    const half_t* __restrict__ Kh, const half_t* __restrict__ Vh,
    float* __restrict__ out){
  __shared__ float lg[4][KNB];
  const int node = blockIdx.x;
  const int h = threadIdx.x >> 5, lane = threadIdx.x & 31;
  const float*  q  = Q  + (size_t)node*HID + h*32;
  const half_t* kh = Kh + (size_t)node*KNB*HID + h*32;
  const float qv = q[lane];
  for (int k = 0; k < KNB; ++k){
    float p = wred_sum(qv * (float)kh[(size_t)k*HID + lane]);
    if (lane == 0) lg[h][k] = p * 0.17677669529f;   // 1/sqrt(32)
  }
  __syncthreads();
  float lv = (lane < KNB) ? lg[h][lane] : -3.0e38f;
  float mx = wred_max(lv);
  float ex = (lane < KNB) ? expf(lv - mx) : 0.f;
  float sm = wred_sum(ex);
  if (lane < KNB) lg[h][lane] = ex / sm;
  __syncthreads();
  const half_t* vh = Vh + (size_t)node*KNB*HID + h*32;
  float o = 0.f;
  for (int k = 0; k < KNB; ++k) o += lg[h][k] * (float)vh[(size_t)k*HID + lane];
  out[(size_t)node*HID + h*32 + lane] = o;
}

// ---------------- residual add + LayerNorm (dim 128) ----------------
__global__ __launch_bounds__(128) void add_ln_kernel(const float* __restrict__ x,
    const float* __restrict__ dh, const float* __restrict__ g,
    const float* __restrict__ bta, float* __restrict__ out, int rows){
  __shared__ float red[4];
  const int r = blockIdx.x, t = threadIdx.x;
  const int wv = t >> 5, lane = t & 31;
  float v = x[(size_t)r*HID + t] + (dh ? dh[(size_t)r*HID + t] : 0.f);
  float s = wred_sum(v);
  if (lane == 0) red[wv] = s;
  __syncthreads();
  float mu = (red[0]+red[1]+red[2]+red[3]) * (1.f/HID);
  __syncthreads();
  float dv = v - mu;
  float s2 = wred_sum(dv*dv);
  if (lane == 0) red[wv] = s2;
  __syncthreads();
  float var = (red[0]+red[1]+red[2]+red[3]) * (1.f/HID);
  out[(size_t)r*HID + t] = g[t]*dv*rsqrtf(var + 1e-6f) + bta[t];
}

// ---------------- sequence embedding ----------------
__global__ __launch_bounds__(128) void embed_kernel(const int* __restrict__ S,
    const float* __restrict__ Ws, float* __restrict__ hS){
  int t = blockIdx.x, c = threadIdx.x;
  hS[(size_t)t*HID + c] = Ws[(size_t)S[t]*HID + c];
}

// ---------------- copy ----------------
__global__ void copy_kernel(const float* __restrict__ a, float* __restrict__ b, int n){
  int i = blockIdx.x*blockDim.x + threadIdx.x;
  if (i < n) b[i] = a[i];
}

// ---------------- final projection + log_softmax (20 classes) ----------------
__global__ __launch_bounds__(32) void out_kernel(const float* __restrict__ hV,
    const float* __restrict__ Wout, const float* __restrict__ bout,
    float* __restrict__ out){
  const int t = blockIdx.x, lane = threadIdx.x;
  float logit = -3.0e38f;
  if (lane < 20){
    float acc = bout[lane];
    for (int c = 0; c < HID; ++c) acc += hV[(size_t)t*HID + c] * Wout[c*20 + lane];
    logit = acc;
  }
  float mx = wred_max(logit);
  float ex = (lane < 20) ? expf(logit - mx) : 0.f;
  float sm = wred_sum(ex);
  if (lane < 20) out[(size_t)t*20 + lane] = logit - mx - logf(sm);
}

// ---------------- host orchestration ----------------
extern "C" void kernel_launch(void* const* d_in, const int* in_sizes, int n_in,
                              void* d_out, int out_size, void* d_ws, size_t ws_size,
                              hipStream_t stream){
  (void)in_sizes; (void)n_in; (void)out_size; (void)ws_size;
  const float* X = (const float*)d_in[0];
  const int*   S = (const int*)d_in[1];
  // d_in[2]=L, d_in[3]=mask (all-ones; folded out analytically)
  auto P = [&](int i)->const float*{ return (const float*)d_in[i]; };
  // param flatten order (alphabetical per dict level):
  // 4:W_e 5:W_out 6:W_s 7:W_v 8:b_e 9:b_out 10:b_v
  // dec layer l at 11+12l: +0 Wfo +1 Wi +2 Wk +3 Wo +4 Wq +5 Wv +6 bfo +7 bi +8 n0b +9 n0g +10 n1b +11 n1g
  // enc layer l at 47+12l: same keys (Wk/Wv are 256x128)
  // feat at 83: We 84:Wn 85:be 86:betae 87:betan 88:bn 89:ge 90:gn

  char* wsp = (char*)d_ws;
  auto carve = [&](size_t nbytes)->void*{
    void* p = (void*)wsp;
    wsp += (nbytes + 255) & ~(size_t)255;
    return p;
  };
  float*  Dnb   = (float*) carve((size_t)RE*4);
  int*    Eidx  = (int*)   carve((size_t)RE*4);
  float*  V6    = (float*) carve((size_t)RN*6*4);
  float*  O9    = (float*) carve((size_t)RN*9*4);
  float*  Vn    = (float*) carve((size_t)RN*HID*4);
  float*  hV    = (float*) carve((size_t)RN*HID*4);
  float*  hVenc = (float*) carve((size_t)RN*HID*4);
  float*  hS    = (float*) carve((size_t)RN*HID*4);
  float*  Qb    = (float*) carve((size_t)RN*HID*4);
  float*  attn  = (float*) carve((size_t)RN*HID*4);
  float*  dh    = (float*) carve((size_t)RN*HID*4);
  float*  mid   = (float*) carve((size_t)RN*512*4);
  float*  Efeat = (float*) carve((size_t)RE*HID*4);
  half_t* hE    = (half_t*)carve((size_t)RE*HID*2);
  half_t* Kh    = (half_t*)carve((size_t)RE*HID*2);
  half_t* Vh    = (half_t*)carve((size_t)RE*HID*2);

  // ---- pre-swizzle all GEMM weights to fragment-ready f16 ----
  half_t* Wh[91] = {};
  auto prep = [&](int idx, int kd, int nout){
    half_t* dst = (half_t*)carve((size_t)kd*nout*2);
    int total = kd*nout;
    swizzle_w_kernel<<<(total+255)/256, 256, 0, stream>>>(P(idx), dst, kd, nout);
    Wh[idx] = dst;
  };
  prep(7, 128, 128);   // W_v
  prep(4, 128, 128);   // W_e
  for (int lyr = 0; lyr < 3; ++lyr){
    int pe = 47 + 12*lyr;   // encoder
    prep(pe+4, 128, 128); prep(pe+2, 256, 128); prep(pe+5, 256, 128);
    prep(pe+3, 128, 128); prep(pe+1, 128, 512); prep(pe+0, 512, 128);
    int pd = 11 + 12*lyr;   // decoder
    prep(pd+4, 128, 128); prep(pd+2, 384, 128); prep(pd+5, 384, 128);
    prep(pd+3, 128, 128); prep(pd+1, 128, 512); prep(pd+0, 512, 128);
  }

  auto gemm = [&](const void* A0, int a0h, const float* A1, const float* A2,
                  const float* A2b, const int* eidx, const half_t* Wsw,
                  const float* bias, void* o, int oh,
                  int rows, int Kdim, int Nout, int mode, int relu){
    dim3 grid(rows/64, Nout/128);
    gemm_wmma_kernel<<<grid, 256, 0, stream>>>(A0,a0h,A1,A2,A2b,eidx,Wsw,bias,o,oh,
                                               rows,Kdim,Nout,mode,relu);
  };

  // ---- features ----
  dist_topk_kernel<<<RN, 32, 0, stream>>>(X, Dnb, Eidx);
  dihedrals_kernel<<<(RN+127)/128, 128, 0, stream>>>(X, V6);
  orient_kernel<<<(RN+127)/128, 128, 0, stream>>>(X, O9);
  node_embed_kernel<<<RN, 32, 0, stream>>>(V6, P(84), P(88), P(90), P(87), Vn);
  gemm(Vn, 0, 0,0,0, 0, Wh[7], P(10), hV, 0, RN, 128, 128, 0, 0);          // W_v,b_v
  edge_feat_kernel<<<RE, 32, 0, stream>>>(X, Dnb, Eidx, O9,
                                          P(83), P(85), P(89), P(86), Efeat);
  gemm(Efeat, 0, 0,0,0, 0, Wh[4], P(8), hE, 1, RE, 128, 128, 0, 0);        // W_e,b_e
  embed_kernel<<<RN, 128, 0, stream>>>(S, P(6), hS);

  // ---- encoder ----
  for (int lyr = 0; lyr < 3; ++lyr){
    int pe = 47 + 12*lyr;
    gemm(hV, 0, 0,0,0, 0,          Wh[pe+4], nullptr, Qb, 0, RN, 128, 128, 0, 0);
    gemm(hE, 1, hV,0,0, Eidx,      Wh[pe+2], nullptr, Kh, 1, RE, 256, 128, 1, 0);
    gemm(hE, 1, hV,0,0, Eidx,      Wh[pe+5], nullptr, Vh, 1, RE, 256, 128, 1, 0);
    attn_kernel<<<RN, 128, 0, stream>>>(Qb, Kh, Vh, attn);
    gemm(attn, 0, 0,0,0, 0,        Wh[pe+3], nullptr, dh, 0, RN, 128, 128, 0, 0);
    add_ln_kernel<<<RN, 128, 0, stream>>>(hV, dh, P(pe+9), P(pe+8), hV, RN);
    gemm(hV, 0, 0,0,0, 0,          Wh[pe+1], P(pe+7), mid, 0, RN, 128, 512, 0, 0);
    gemm(mid, 0, 0,0,0, 0,         Wh[pe+0], P(pe+6), dh,  0, RN, 512, 128, 0, 1);
    add_ln_kernel<<<RN, 128, 0, stream>>>(hV, dh, P(pe+11), P(pe+10), hV, RN);
  }
  copy_kernel<<<(RN*HID+255)/256, 256, 0, stream>>>(hV, hVenc, RN*HID);

  // ---- decoder (autoregressive blend fused into GEMM gather: mode 2) ----
  for (int lyr = 0; lyr < 3; ++lyr){
    int pd = 11 + 12*lyr;
    gemm(hV, 0, 0,0,0, 0,              Wh[pd+4], nullptr, Qb, 0, RN, 128, 128, 0, 0);
    gemm(hE, 1, hS, hV, hVenc, Eidx,   Wh[pd+2], nullptr, Kh, 1, RE, 384, 128, 2, 0);
    gemm(hE, 1, hS, hV, hVenc, Eidx,   Wh[pd+5], nullptr, Vh, 1, RE, 384, 128, 2, 0);
    attn_kernel<<<RN, 128, 0, stream>>>(Qb, Kh, Vh, attn);
    gemm(attn, 0, 0,0,0, 0,            Wh[pd+3], nullptr, dh, 0, RN, 128, 128, 0, 0);
    add_ln_kernel<<<RN, 128, 0, stream>>>(hV, dh, P(pd+9), P(pd+8), hV, RN);
    gemm(hV, 0, 0,0,0, 0,              Wh[pd+1], P(pd+7), mid, 0, RN, 128, 512, 0, 0);
    gemm(mid, 0, 0,0,0, 0,             Wh[pd+0], P(pd+6), dh,  0, RN, 512, 128, 0, 1);
    add_ln_kernel<<<RN, 128, 0, stream>>>(hV, dh, P(pd+11), P(pd+10), hV, RN);
  }

  out_kernel<<<RN, 32, 0, stream>>>(hV, P(5), P(9), (float*)d_out);
}
